// InvertedResidualBlock_27994596835974
// MI455X (gfx1250) — compile-verified, matrix operand order checked by reading, not run on hardware
//
#include <hip/hip_runtime.h>
#include <hip/hip_bf16.h>

typedef _Float16 f16;
typedef _Float16 v8h  __attribute__((ext_vector_type(8)));
typedef _Float16 v16h __attribute__((ext_vector_type(16)));
typedef float    v8f  __attribute__((ext_vector_type(8)));

#define BATCH 32
#define CIN   64
#define HW    56
#define NPIX  (HW*HW)          /* 3136   */
#define HID   384
#define RED   16
#define COUT  64
#define MTOT  (BATCH*NPIX)     /* 100352 */
#define KEXP  (9*CIN)          /* 576    */
#define NKE   (KEXP/32)        /* 18     */
#define NKP   (HID/32)         /* 12     */
#define NPT   49
#define BN_EPS 1e-5f
#define LDSP  40               /* GEMM LDS row stride (halves): 80B, 16B-mult */
#define CSTR  72               /* epilogue C-tile stride (halves): 144B, 16B-mult */

// ---- workspace layout (bytes, all 256B aligned) ----
#define OFF_H    0ull            /* f16 h [MTOT][HID]      77,070,336 */
#define OFF_D    77070336ull     /* f16 d [MTOT][HID]      77,070,336 */
#define OFF_BEXP 154140672ull    /* f16 BexpP [6][18][64][32] 442,368 */
#define OFF_BPW  154583040ull    /* f16 BpwP  [12][64][32]     49,152 */
#define OFF_SC1  154632192ull
#define OFF_SC2  154635264ull
#define OFF_SC3  154638336ull
#define OFF_POOL 154638848ull
#define OFF_S2   154688000ull    /* f16 gate [32][384]                */
#define OFF_WDWT 154737152ull    /* f16 wdwT [9][384]                 */
#define OFF_XH   154744320ull    /* f16 xh [MTOT][64]      12,845,056 */
#define OFF_PART 167589376ull    /* f32 part [32][49][384]  2,408,448 */

__device__ __forceinline__ float silu_f(float v) { return v * (1.0f/(1.0f + __expf(-v))); }
__device__ __forceinline__ float sigm_f(float v) { return 1.0f/(1.0f + __expf(-v)); }

// ---------------- prep: expand weights -> staged-tile order ------------------------
// BexpP[((nblk*NKE + ks)*64 + n)*32 + k] = w_exp[nblk*64+n][c][kh][kw],
// where kglob = ks*32+k, tap = kglob>>6, c = kglob&63.
__global__ __launch_bounds__(256) void k_pack_exp(const float* __restrict__ w, f16* __restrict__ BexpP) {
    int i = blockIdx.x*256 + threadIdx.x;
    if (i >= KEXP*HID) return;
    int k  = i & 31;
    int r  = i >> 5;
    int n  = r & 63;
    int r2 = r >> 6;                 // nblk*NKE + ks
    int ks   = r2 % NKE;
    int nblk = r2 / NKE;
    int kglob = ks*32 + k;
    int tap = kglob >> 6, c = kglob & 63;
    int ng  = nblk*64 + n;
    BexpP[i] = (f16)w[(ng*CIN + c)*9 + tap];
}

// ---------------- prep: pointwise weights -> staged-tile order ---------------------
// BpwP[((ks)*64 + n)*32 + k] = w_pw[n][ks*32+k]
__global__ __launch_bounds__(256) void k_pack_pw(const float* __restrict__ w, f16* __restrict__ BpwP) {
    int i = blockIdx.x*256 + threadIdx.x;
    if (i >= HID*COUT) return;
    int k  = i & 31;
    int r  = i >> 5;
    int n  = r & 63;
    int ks = r >> 6;
    BpwP[i] = (f16)w[n*HID + ks*32 + k];
}

__global__ __launch_bounds__(256) void k_pack_dw(const float* __restrict__ w, f16* __restrict__ wdwT) {
    int i = blockIdx.x*256 + threadIdx.x;
    if (i >= 9*HID) return;
    int tap = i / HID, c = i - tap*HID;
    wdwT[tap*HID + c] = (f16)w[c*9 + tap];
}

__global__ __launch_bounds__(256) void k_bn_prep(
    const float* g1,const float* b1,const float* m1,const float* v1,
    const float* g2,const float* b2,const float* m2,const float* v2,
    const float* g3,const float* b3,const float* m3,const float* v3,
    float* sc1, float* sc2, float* sc3)
{
    int i = blockIdx.x*256 + threadIdx.x;
    if (i < HID) {
        float s = g1[i]*rsqrtf(v1[i]+BN_EPS);
        sc1[i] = s; sc1[HID+i] = b1[i] - m1[i]*s;
    } else if (i < 2*HID) {
        int c = i - HID;
        float s = g2[c]*rsqrtf(v2[c]+BN_EPS);
        sc2[c] = s; sc2[HID+c] = b2[c] - m2[c]*s;
    } else if (i < 2*HID + COUT) {
        int c = i - 2*HID;
        float s = g3[c]*rsqrtf(v3[c]+BN_EPS);
        sc3[c] = s; sc3[COUT+c] = b3[c] - m3[c]*s;
    }
}

// ---------------- x: NCHW f32 -> NHWC f16 ------------------------------------------
__global__ __launch_bounds__(256) void k_tr_x(const float* __restrict__ x, f16* __restrict__ xh) {
    __shared__ __align__(16) f16 tileT[64*80];
    const int b  = blockIdx.x / NPT;
    const int pt = blockIdx.x - b*NPT;
    const int p0 = pt*64;
    const int t  = threadIdx.x;
    #pragma unroll
    for (int j = 0; j < 16; ++j) {
        const int idx = j*256 + t;
        const int c = idx >> 6, p = idx & 63;
        tileT[p*80 + c] = (f16)x[(size_t)(b*CIN + c)*NPIX + p0 + p];
    }
    __syncthreads();
    #pragma unroll
    for (int j = 0; j < 2; ++j) {
        const int idx = j*256 + t;
        const int p = idx >> 3, cg = (idx & 7)*8;
        v8h v = *reinterpret_cast<const v8h*>(&tileT[p*80 + cg]);
        *reinterpret_cast<v8h*>(&xh[(size_t)(b*NPIX + p0 + p)*CIN + cg]) = v;
    }
}

// ---------------- expand conv 3x3: implicit GEMM, WMMA f16 -------------------------
// grid (392, 6), block 256 = 8 waves; tile 256M x 64N; wave: 32M x 64N (8 acc tiles)
__global__ __launch_bounds__(256) void k_expand_wmma(
    const f16* __restrict__ xh, const f16* __restrict__ BexpP,
    const float* __restrict__ sc1, f16* __restrict__ h)
{
    __shared__ __align__(16) f16 Alds[2][256*LDSP];     // 40,960 B
    __shared__ __align__(16) f16 Blds[2][64*LDSP];      // 10,240 B
    const int tid  = threadIdx.x;
    const int lane = tid & 31;
    const int wave = tid >> 5;
    const int mblk = blockIdx.x;
    const int nblk = blockIdx.y;
    const int n0   = nblk * 64;

    // A staging: thread owns one of 256 rows; coords constant across K-steps
    const int mgS  = mblk*256 + tid;
    const int bS   = mgS / NPIX;
    const int pixS = mgS - bS*NPIX;
    const int yS   = pixS / HW;
    const int xS   = pixS - yS*HW;
    const f16* baseP = xh + (size_t)(bS*NPIX + yS*HW + xS)*CIN;
    unsigned bmask = 0;
    #pragma unroll
    for (int tap = 0; tap < 9; ++tap) {
        const int iy = yS + tap/3 - 1, ix = xS + tap%3 - 1;
        if ((unsigned)iy < (unsigned)HW && (unsigned)ix < (unsigned)HW) bmask |= (1u << tap);
    }
    // B staging: whole 64x32 tile is contiguous; thread owns one b128 slice
    const int nB  = tid >> 2;            // 0..63
    const int koB = (tid & 3) * 8;       // 0,8,16,24
    const f16* bTiles = BexpP + ((size_t)(nblk*NKE)*64 + nB)*32 + koB;

    v8f acc[2][4];
    #pragma unroll
    for (int p = 0; p < 2; ++p)
        #pragma unroll
        for (int t = 0; t < 4; ++t) acc[p][t] = (v8f)0.0f;

    const int m0   = wave * 32;
    const int klo  = (lane < 16) ? 0 : 8;
    const int arow0 = m0 + (lane & 15);
    const int arow1 = arow0 + 16;
    const int bcol = lane & 15;

    #define STAGE_EXP(KS, BUF)                                                         \
    {                                                                                  \
        const int tap = (KS) >> 1;                                                     \
        const int c0  = ((KS) & 1) * 32;                                               \
        const int toff = ((tap/3 - 1)*HW + (tap%3 - 1))*CIN + c0;                      \
        v8h a0 = (v8h)(f16)0.0f, a1 = a0, a2 = a0, a3 = a0;                            \
        if ((bmask >> tap) & 1u) {                                                     \
            const v8h* p = reinterpret_cast<const v8h*>(baseP + toff);                 \
            a0 = p[0]; a1 = p[1]; a2 = p[2]; a3 = p[3];                                \
        }                                                                              \
        f16* ad = &Alds[BUF][tid*LDSP];                                                \
        *reinterpret_cast<v8h*>(ad)      = a0;                                         \
        *reinterpret_cast<v8h*>(ad + 8)  = a1;                                         \
        *reinterpret_cast<v8h*>(ad + 16) = a2;                                         \
        *reinterpret_cast<v8h*>(ad + 24) = a3;                                         \
        v8h bv = *reinterpret_cast<const v8h*>(bTiles + (size_t)(KS)*64*32);           \
        *reinterpret_cast<v8h*>(&Blds[BUF][nB*LDSP + koB]) = bv;                       \
    }

    STAGE_EXP(0, 0);
    __syncthreads();
    #pragma unroll
    for (int ks = 0; ks < NKE; ++ks) {
        const int cur = ks & 1;
        if (ks + 1 < NKE) { STAGE_EXP(ks + 1, cur ^ 1); }
        v8h al0 = *reinterpret_cast<const v8h*>(&Alds[cur][arow0*LDSP + klo]);
        v8h ah0 = *reinterpret_cast<const v8h*>(&Alds[cur][arow0*LDSP + klo + 16]);
        v8h al1 = *reinterpret_cast<const v8h*>(&Alds[cur][arow1*LDSP + klo]);
        v8h ah1 = *reinterpret_cast<const v8h*>(&Alds[cur][arow1*LDSP + klo + 16]);
        v16h af0 = __builtin_shufflevector(al0, ah0, 0,1,2,3,4,5,6,7,8,9,10,11,12,13,14,15);
        v16h af1 = __builtin_shufflevector(al1, ah1, 0,1,2,3,4,5,6,7,8,9,10,11,12,13,14,15);
        #pragma unroll
        for (int t = 0; t < 4; ++t) {
            const int col = t*16 + bcol;
            v8h blo = *reinterpret_cast<const v8h*>(&Blds[cur][col*LDSP + klo]);
            v8h bhi = *reinterpret_cast<const v8h*>(&Blds[cur][col*LDSP + klo + 16]);
            v16h bf = __builtin_shufflevector(blo, bhi, 0,1,2,3,4,5,6,7,8,9,10,11,12,13,14,15);
            acc[0][t] = __builtin_amdgcn_wmma_f32_16x16x32_f16(
                false, af0, false, bf, (short)0, acc[0][t], false, false);
            acc[1][t] = __builtin_amdgcn_wmma_f32_16x16x32_f16(
                false, af1, false, bf, (short)0, acc[1][t], false, false);
        }
        __syncthreads();
    }
    #undef STAGE_EXP

    // epilogue: BN1 + SiLU, transpose through LDS, coalesced b128 stores
    f16* Ct = (f16*)Alds;                                // 256 x CSTR = 18,432 halves
    const int hiM = (lane >= 16) ? 8 : 0;
    #pragma unroll
    for (int p = 0; p < 2; ++p) {
        const int mb = m0 + p*16 + hiM;
        #pragma unroll
        for (int t = 0; t < 4; ++t) {
            const int chl = t*16 + (lane & 15);
            const float s  = sc1[n0 + chl];
            const float sh = sc1[HID + n0 + chl];
            #pragma unroll
            for (int r = 0; r < 8; ++r)
                Ct[(mb + r)*CSTR + chl] = (f16)silu_f(acc[p][t][r]*s + sh);
        }
    }
    __syncthreads();
    {
        const f16* src = &Ct[tid*CSTR];
        f16* dst = &h[(size_t)(mblk*256 + tid)*HID + n0];
        #pragma unroll
        for (int q = 0; q < 8; ++q)
            *reinterpret_cast<v8h*>(dst + q*8) = *reinterpret_cast<const v8h*>(src + q*8);
    }
}

// ---------------- depthwise 3x3 + BN2 + SiLU ---------------------------------------
__global__ __launch_bounds__(256) void k_dw(
    const f16* __restrict__ h, const f16* __restrict__ wdwT,
    const float* __restrict__ sc2, f16* __restrict__ d)
{
    const int idx = blockIdx.x*256 + threadIdx.x;        // MTOT*48 exact
    const int m  = idx / 48;
    const int c0 = (idx - m*48) * 8;
    const int b   = m / NPIX;
    const int pix = m - b*NPIX;
    const int yy  = pix / HW, xx = pix - yy*HW;
    float acc[8];
    #pragma unroll
    for (int j = 0; j < 8; ++j) acc[j] = 0.0f;

    if (yy >= 1 && yy < HW-1 && xx >= 1 && xx < HW-1) {  // interior fast path (93%)
        const f16* hp = &h[(size_t)(b*NPIX + (yy-1)*HW + (xx-1))*HID + c0];
        #pragma unroll
        for (int tap = 0; tap < 9; ++tap) {
            v8h hv = *reinterpret_cast<const v8h*>(&hp[(size_t)((tap/3)*HW + tap%3)*HID]);
            v8h wv = *reinterpret_cast<const v8h*>(&wdwT[tap*HID + c0]);
            #pragma unroll
            for (int j = 0; j < 8; ++j) acc[j] += (float)hv[j]*(float)wv[j];
        }
    } else {
        #pragma unroll
        for (int tap = 0; tap < 9; ++tap) {
            const int iy = yy + tap/3 - 1;
            const int ix = xx + tap%3 - 1;
            if ((unsigned)iy < (unsigned)HW && (unsigned)ix < (unsigned)HW) {
                v8h hv = *reinterpret_cast<const v8h*>(&h[(size_t)(b*NPIX + iy*HW + ix)*HID + c0]);
                v8h wv = *reinterpret_cast<const v8h*>(&wdwT[tap*HID + c0]);
                #pragma unroll
                for (int j = 0; j < 8; ++j) acc[j] += (float)hv[j]*(float)wv[j];
            }
        }
    }
    v8h o;
    #pragma unroll
    for (int j = 0; j < 8; ++j)
        o[j] = (f16)silu_f(acc[j]*sc2[c0+j] + sc2[HID+c0+j]);
    *reinterpret_cast<v8h*>(&d[(size_t)m*HID + c0]) = o;
}

// ---------------- SE pool (two-stage, deterministic) -------------------------------
__global__ __launch_bounds__(384) void k_pool1(const f16* __restrict__ d, float* __restrict__ part) {
    const int b = blockIdx.x, pt = blockIdx.y;
    const int c = threadIdx.x;
    const int p0 = pt*64;
    float s = 0.0f;
    for (int p = 0; p < 64; ++p)
        s += (float)d[(size_t)(b*NPIX + p0 + p)*HID + c];
    part[(size_t)(b*NPT + pt)*HID + c] = s;
}
__global__ __launch_bounds__(256) void k_pool2(const float* __restrict__ part, float* __restrict__ pool) {
    const int i = blockIdx.x*256 + threadIdx.x;          // BATCH*HID exact
    const int b = i / HID, c = i - b*HID;
    float s = 0.0f;
    for (int pt = 0; pt < NPT; ++pt)
        s += part[(size_t)(b*NPT + pt)*HID + c];
    pool[i] = s * (1.0f/(float)NPIX);
}

// ---------------- SE MLP -----------------------------------------------------------
__global__ __launch_bounds__(256) void k_se(
    const float* __restrict__ pool, const float* __restrict__ w1,
    const float* __restrict__ w2, f16* __restrict__ gate)
{
    __shared__ float meanv[HID];
    __shared__ float s1[RED];
    const int b = blockIdx.x, tid = threadIdx.x;
    for (int c = tid; c < HID; c += 256) meanv[c] = pool[b*HID + c];
    __syncthreads();
    if (tid < RED) {
        float a = 0.0f;
        for (int c = 0; c < HID; ++c) a += meanv[c]*w1[tid*HID + c];
        s1[tid] = silu_f(a);
    }
    __syncthreads();
    for (int c = tid; c < HID; c += 256) {
        float a = 0.0f;
        #pragma unroll
        for (int r = 0; r < RED; ++r) a += s1[r]*w2[c*RED + r];
        gate[b*HID + c] = (f16)sigm_f(a);
    }
}

// ---------------- pointwise GEMM (WMMA) + BN3 + residual ---------------------------
// grid 392, block 256 = 8 waves; tile 256M x 64N; wave: 32M x 64N; K=384 unrolled.
__global__ __launch_bounds__(256) void k_pw_wmma(
    const f16* __restrict__ d, const f16* __restrict__ gate,
    const f16* __restrict__ BpwP, const float* __restrict__ sc3,
    const float* __restrict__ x, float* __restrict__ out)
{
    __shared__ __align__(16) f16 Alds[2][256*LDSP];
    __shared__ __align__(16) f16 Blds[2][64*LDSP];
    const int tid  = threadIdx.x;
    const int lane = tid & 31;
    const int wave = tid >> 5;
    const int mblk = blockIdx.x;

    const int mgS = mblk*256 + tid;
    const int bS  = mgS / NPIX;
    const int gRow = bS*HID;
    const int nB  = tid >> 2;
    const int koB = (tid & 3) * 8;
    const f16* bTiles = BpwP + (size_t)nB*32 + koB;

    v8f acc[2][4];
    #pragma unroll
    for (int p = 0; p < 2; ++p)
        #pragma unroll
        for (int t = 0; t < 4; ++t) acc[p][t] = (v8f)0.0f;

    const int m0   = wave * 32;
    const int klo  = (lane < 16) ? 0 : 8;
    const int arow0 = m0 + (lane & 15);
    const int arow1 = arow0 + 16;
    const int bcol = lane & 15;

    #define STAGE_PW(KS, BUF)                                                          \
    {                                                                                  \
        const int kbase = (KS)*32;                                                     \
        const v8h* dp = reinterpret_cast<const v8h*>(&d[(size_t)mgS*HID + kbase]);     \
        const v8h* gp = reinterpret_cast<const v8h*>(&gate[gRow + kbase]);             \
        f16* ad = &Alds[BUF][tid*LDSP];                                                \
        _Pragma("unroll")                                                              \
        for (int q = 0; q < 4; ++q)                                                    \
            *reinterpret_cast<v8h*>(ad + q*8) = dp[q] * gp[q];                         \
        v8h bv = *reinterpret_cast<const v8h*>(bTiles + (size_t)(KS)*64*32);           \
        *reinterpret_cast<v8h*>(&Blds[BUF][nB*LDSP + koB]) = bv;                       \
    }

    STAGE_PW(0, 0);
    __syncthreads();
    #pragma unroll
    for (int ks = 0; ks < NKP; ++ks) {
        const int cur = ks & 1;
        if (ks + 1 < NKP) { STAGE_PW(ks + 1, cur ^ 1); }
        v8h al0 = *reinterpret_cast<const v8h*>(&Alds[cur][arow0*LDSP + klo]);
        v8h ah0 = *reinterpret_cast<const v8h*>(&Alds[cur][arow0*LDSP + klo + 16]);
        v8h al1 = *reinterpret_cast<const v8h*>(&Alds[cur][arow1*LDSP + klo]);
        v8h ah1 = *reinterpret_cast<const v8h*>(&Alds[cur][arow1*LDSP + klo + 16]);
        v16h af0 = __builtin_shufflevector(al0, ah0, 0,1,2,3,4,5,6,7,8,9,10,11,12,13,14,15);
        v16h af1 = __builtin_shufflevector(al1, ah1, 0,1,2,3,4,5,6,7,8,9,10,11,12,13,14,15);
        #pragma unroll
        for (int t = 0; t < 4; ++t) {
            const int col = t*16 + bcol;
            v8h blo = *reinterpret_cast<const v8h*>(&Blds[cur][col*LDSP + klo]);
            v8h bhi = *reinterpret_cast<const v8h*>(&Blds[cur][col*LDSP + klo + 16]);
            v16h bf = __builtin_shufflevector(blo, bhi, 0,1,2,3,4,5,6,7,8,9,10,11,12,13,14,15);
            acc[0][t] = __builtin_amdgcn_wmma_f32_16x16x32_f16(
                false, af0, false, bf, (short)0, acc[0][t], false, false);
            acc[1][t] = __builtin_amdgcn_wmma_f32_16x16x32_f16(
                false, af1, false, bf, (short)0, acc[1][t], false, false);
        }
        __syncthreads();
    }
    #undef STAGE_PW

    // epilogue: BN3 + residual; 8 acc rows = 8 consecutive pixels (3136 % 8 == 0)
    const int hiM = (lane >= 16) ? 8 : 0;
    #pragma unroll
    for (int p = 0; p < 2; ++p) {
        const int mg0 = mblk*256 + m0 + p*16 + hiM;      // multiple of 8
        const int bb  = mg0 / NPIX;
        const int pix = mg0 - bb*NPIX;
        #pragma unroll
        for (int t = 0; t < 4; ++t) {
            const int co = t*16 + (lane & 15);
            const float s  = sc3[co];
            const float sh = sc3[COUT + co];
            const size_t oi = (size_t)(bb*COUT + co)*NPIX + pix;
            float4 x0 = *reinterpret_cast<const float4*>(&x[oi]);
            float4 x1 = *reinterpret_cast<const float4*>(&x[oi + 4]);
            float4 o0, o1;
            o0.x = acc[p][t][0]*s + sh + x0.x;  o0.y = acc[p][t][1]*s + sh + x0.y;
            o0.z = acc[p][t][2]*s + sh + x0.z;  o0.w = acc[p][t][3]*s + sh + x0.w;
            o1.x = acc[p][t][4]*s + sh + x1.x;  o1.y = acc[p][t][5]*s + sh + x1.y;
            o1.z = acc[p][t][6]*s + sh + x1.z;  o1.w = acc[p][t][7]*s + sh + x1.w;
            *reinterpret_cast<float4*>(&out[oi])     = o0;
            *reinterpret_cast<float4*>(&out[oi + 4]) = o1;
        }
    }
}

extern "C" void kernel_launch(void* const* d_in, const int* in_sizes, int n_in,
                              void* d_out, int out_size, void* d_ws, size_t ws_size,
                              hipStream_t stream)
{
    const float* x    = (const float*)d_in[0];
    const float* wexp = (const float*)d_in[1];
    const float* g1 = (const float*)d_in[2],  *b1 = (const float*)d_in[3];
    const float* m1 = (const float*)d_in[4],  *v1 = (const float*)d_in[5];
    const float* wdw  = (const float*)d_in[6];
    const float* g2 = (const float*)d_in[7],  *b2 = (const float*)d_in[8];
    const float* m2 = (const float*)d_in[9],  *v2 = (const float*)d_in[10];
    const float* wse1 = (const float*)d_in[11];
    const float* wse2 = (const float*)d_in[12];
    const float* wpw  = (const float*)d_in[13];
    const float* g3 = (const float*)d_in[14], *b3 = (const float*)d_in[15];
    const float* m3 = (const float*)d_in[16], *v3 = (const float*)d_in[17];
    float* out = (float*)d_out;

    char* ws = (char*)d_ws;
    f16*   h    = (f16*)  (ws + OFF_H);
    f16*   dbuf = (f16*)  (ws + OFF_D);
    f16*   BexpP= (f16*)  (ws + OFF_BEXP);
    f16*   BpwP = (f16*)  (ws + OFF_BPW);
    float* sc1  = (float*)(ws + OFF_SC1);
    float* sc2  = (float*)(ws + OFF_SC2);
    float* sc3  = (float*)(ws + OFF_SC3);
    float* pool = (float*)(ws + OFF_POOL);
    f16*   gate = (f16*)  (ws + OFF_S2);
    f16*   wdwT = (f16*)  (ws + OFF_WDWT);
    f16*   xh   = (f16*)  (ws + OFF_XH);
    float* part = (float*)(ws + OFF_PART);

    k_pack_exp<<<(KEXP*HID + 255)/256, 256, 0, stream>>>(wexp, BexpP);
    k_pack_pw <<<(HID*COUT + 255)/256, 256, 0, stream>>>(wpw, BpwP);
    k_pack_dw <<<(9*HID + 255)/256,    256, 0, stream>>>(wdw, wdwT);
    k_bn_prep <<<4, 256, 0, stream>>>(g1,b1,m1,v1, g2,b2,m2,v2, g3,b3,m3,v3, sc1,sc2,sc3);

    k_tr_x<<<BATCH*NPT, 256, 0, stream>>>(x, xh);

    dim3 gexp(MTOT/256, HID/64);                          // 392 x 6
    k_expand_wmma<<<gexp, 256, 0, stream>>>(xh, BexpP, sc1, h);

    k_dw<<<(MTOT*48)/256, 256, 0, stream>>>(h, wdwT, sc2, dbuf);

    k_pool1<<<dim3(BATCH, NPT), 384, 0, stream>>>(dbuf, part);
    k_pool2<<<(BATCH*HID)/256, 256, 0, stream>>>(part, pool);
    k_se<<<BATCH, 256, 0, stream>>>(pool, wse1, wse2, gate);

    k_pw_wmma<<<MTOT/256, 256, 0, stream>>>(dbuf, gate, BpwP, sc3, x, out);
}
